// eNest_82282983457299
// MI455X (gfx1250) — matrix operand order repaired and checked by reading, not compiled, not run on hardware
//
#include <hip/hip_runtime.h>
#include <math.h>

// ---------------------------------------------------------------------------
// Problem constants (from the reference)
// ---------------------------------------------------------------------------
#define BN_   8192      // batch
#define G_    689       // genes
#define S_    2785      // layer width
#define L_    6         // scanned layers
#define SP_   2816      // S padded to multiple of 64 (and 32 for WMMA K)
#define KP0_  704       // G padded to multiple of 32

typedef __attribute__((ext_vector_type(16))) __bf16 v16bf;
typedef __attribute__((ext_vector_type(8)))  float  v8f;

__device__ __forceinline__ unsigned short f2bf(float f) {
    // round-to-nearest-even float -> bf16 (bit-level, no __bf16 scalar math)
    unsigned int u = __float_as_uint(f);
    unsigned int r = 0x7FFFu + ((u >> 16) & 1u);
    return (unsigned short)((u + r) >> 16);
}

// ---------------------------------------------------------------------------
// Masked transpose + fp32->bf16:  out[n*Kp + k] = W[k*ld + n] * M[k*ld + n]
// (zero-padded to Kp x Np).  LDS-tiled so both sides stay coalesced.
// ---------------------------------------------------------------------------
__global__ __launch_bounds__(256) void prep_weights_kernel(
    const float* __restrict__ Wsrc, const float* __restrict__ Msrc,
    unsigned short* __restrict__ out,
    int K, int N, int ld, int Kp,
    long srcStrideZ, long outStrideZ)
{
    __shared__ float tile[32][33];
    const int z = blockIdx.z;
    const float* Wp = Wsrc + (long)z * srcStrideZ;
    const float* Mp = Msrc + (long)z * srcStrideZ;
    unsigned short* op = out + (long)z * outStrideZ;

    const int k0 = blockIdx.x * 32;
    const int n0 = blockIdx.y * 32;
    const int tx = threadIdx.x & 31;
    const int ty = threadIdx.x >> 5;   // 0..7

    #pragma unroll
    for (int r = 0; r < 32; r += 8) {
        int k = k0 + ty + r, n = n0 + tx;
        float v = 0.0f;
        if (k < K && n < N) v = Wp[(long)k * ld + n] * Mp[(long)k * ld + n];
        tile[ty + r][tx] = v;
    }
    __syncthreads();
    #pragma unroll
    for (int r = 0; r < 32; r += 8) {
        int n = n0 + ty + r, k = k0 + tx;
        op[(long)n * Kp + k] = f2bf(tile[tx][ty + r]);
    }
}

// ---------------------------------------------------------------------------
// x [B,G] fp32 -> A0 [B,KP0] bf16 (zero padded)
// ---------------------------------------------------------------------------
__global__ __launch_bounds__(256) void prep_a0_kernel(
    const float* __restrict__ x, unsigned short* __restrict__ A0, long total)
{
    long i = (long)blockIdx.x * 256 + threadIdx.x;
    if (i >= total) return;
    long b = i / KP0_;
    int  k = (int)(i % KP0_);
    A0[i] = (k < G_) ? f2bf(x[b * G_ + k]) : (unsigned short)0;
}

// ---------------------------------------------------------------------------
// WMMA GEMM + fused BN column stats.
//   Z[B,Np] = A[B,Kp](bf16) * WtT[Np,Kp](bf16, N-major/K-contiguous)
// block = 256 threads (8 wave32), block tile 256(M) x 64(N),
// wave tile 32(M) x 64(N) = 8 x v_wmma_f32_16x16x32_bf16 per K-step
// (12 b128 loads feed 8 WMMAs -> ~2x flops per VMEM instr vs a 16x64 tile).
// Epilogue reduces per-column sum / sum-of-squares for the block's 256-row
// slab (shfl_xor across half-wave row groups + LDS cross-wave fold) and
// writes one deterministic partial per slab -> the BN stats pass never has
// to re-read the 92 MB Z tensor from HBM.
// Fragment layouts follow CDNA5 ISA 7.12.2 (16-bit A 16x32 / B 32x16).
// ---------------------------------------------------------------------------
union FragU { v16bf v; uint4 q[2]; };

__global__ __launch_bounds__(256) void gemm_bf16_kernel(
    const unsigned short* __restrict__ A,    // [B, Kp]
    const unsigned short* __restrict__ Wt,   // [Np, Kp]
    float* __restrict__ Z,                   // [B, Np]
    float* __restrict__ psum,                // [B/256, Np]
    float* __restrict__ psumsq,              // [B/256, Np]
    int Kp, int Np)
{
    const int lane = threadIdx.x & 31;
    const int wid  = threadIdx.x >> 5;       // 0..7 -> 32-row M sub-tile
    const int mrow = lane & 15;
    const int half = lane >> 4;              // 0/1: lane group

    const long m_base = (long)blockIdx.y * 256 + (long)wid * 32;
    const long n_base = (long)blockIdx.x * 64;

    v8f acc[2][4] = {};

    const unsigned short* aR[2];
    aR[0] = A + (m_base + mrow) * (long)Kp;
    aR[1] = A + (m_base + 16 + mrow) * (long)Kp;
    const unsigned short* bR[4];
    #pragma unroll
    for (int nt = 0; nt < 4; ++nt)
        bR[nt] = Wt + (n_base + nt * 16 + mrow) * (long)Kp;

    for (int k0 = 0; k0 < Kp; k0 += 32) {
        FragU a[2], b[4];
        // A 16x32 bf16: elems 0..7 = K[half*8 .. +7], elems 8..15 = K[16+half*8 .. +7]
        const int ka = k0 + half * 8;
        #pragma unroll
        for (int mt = 0; mt < 2; ++mt) {
            a[mt].q[0] = *(const uint4*)(aR[mt] + ka);
            a[mt].q[1] = *(const uint4*)(aR[mt] + ka + 16);
        }
        // B 32x16 bf16: elems 0..7 = K[half*8 .. +7], elems 8..15 = K[16+half*8 .. +7] of column n
        const int kb = k0 + half * 8;
        #pragma unroll
        for (int nt = 0; nt < 4; ++nt) {
            b[nt].q[0] = *(const uint4*)(bR[nt] + kb);
            b[nt].q[1] = *(const uint4*)(bR[nt] + kb + 16);
        }
        #pragma unroll
        for (int mt = 0; mt < 2; ++mt)
            #pragma unroll
            for (int nt = 0; nt < 4; ++nt)
                acc[mt][nt] = __builtin_amdgcn_wmma_f32_16x16x32_bf16(
                    false, a[mt].v, false, b[nt].v, (short)0, acc[mt][nt],
                    false, false);
    }

    // ---- store Z.  C/D layout: VGPR r -> row (half*8 + r), lane&15 -> column
    const int ccol = lane & 15;
    #pragma unroll
    for (int mt = 0; mt < 2; ++mt) {
        float* zp = Z + (m_base + mt * 16 + half * 8) * (long)Np + n_base + ccol;
        #pragma unroll
        for (int r = 0; r < 8; ++r) {
            zp[0]  = acc[mt][0][r];
            zp[16] = acc[mt][1][r];
            zp[32] = acc[mt][2][r];
            zp[48] = acc[mt][3][r];
            zp += Np;
        }
    }

    // ---- fused column partial sums over this block's 256 rows
    __shared__ float redS[8][64];
    __shared__ float redQ[8][64];
    #pragma unroll
    for (int nt = 0; nt < 4; ++nt) {
        float s = 0.0f, q = 0.0f;
        #pragma unroll
        for (int mt = 0; mt < 2; ++mt)
            #pragma unroll
            for (int r = 0; r < 8; ++r) {
                float v = acc[mt][nt][r];
                s += v;
                q += v * v;
            }
        // merge the two half-wave row groups (same column, different rows)
        s += __shfl_xor(s, 16, 32);
        q += __shfl_xor(q, 16, 32);
        if (half == 0) {
            redS[wid][nt * 16 + mrow] = s;
            redQ[wid][nt * 16 + mrow] = q;
        }
    }
    __syncthreads();
    if (threadIdx.x < 64) {
        float fs = 0.0f, fq = 0.0f;
        #pragma unroll
        for (int w = 0; w < 8; ++w) {
            fs += redS[w][threadIdx.x];
            fq += redQ[w][threadIdx.x];
        }
        psum  [(long)blockIdx.y * Np + n_base + threadIdx.x] = fs;
        psumsq[(long)blockIdx.y * Np + n_base + threadIdx.x] = fq;
    }
}

// Fold 32 slab partials (fixed order -> deterministic), produce per-column
// BN scale/shift.  (bias b cancels inside batch-mean subtraction -> omitted
// exactly.)
__global__ __launch_bounds__(256) void stats_finalize_kernel(
    const float* __restrict__ psum, const float* __restrict__ psumsq,
    const float* __restrict__ gamma, const float* __restrict__ beta,
    float* __restrict__ scale, float* __restrict__ shift, int Np)
{
    const int col = blockIdx.x * 256 + threadIdx.x;
    if (col >= Np) return;
    if (col < S_) {
        float s = 0.0f, ss = 0.0f;
        for (int p = 0; p < BN_ / 256; ++p) {
            s  += psum  [(long)p * Np + col];
            ss += psumsq[(long)p * Np + col];
        }
        const float invB = 1.0f / (float)BN_;
        float m   = s * invB;
        float var = ss * invB - m * m;
        float sc  = rsqrtf(var + 1e-5f) * gamma[col];
        scale[col] = sc;
        shift[col] = beta[col] - m * sc;
    } else {
        scale[col] = 0.0f;
        shift[col] = 0.0f;
    }
}

// ---------------------------------------------------------------------------
// h_next = tanh(z*scale + shift) [+ h*diag];  writes fp32 H (in place) and
// the bf16 activation copy for the next GEMM's A operand.
// ---------------------------------------------------------------------------
__global__ __launch_bounds__(256) void apply_kernel(
    const float* __restrict__ Z, float* __restrict__ H,
    unsigned short* __restrict__ Abf,
    const float* __restrict__ scale, const float* __restrict__ shift,
    const float* __restrict__ diag, int use_skip, long total)
{
    long i = (long)blockIdx.x * 256 + threadIdx.x;
    if (i >= total) return;
    const int col = (int)(i % SP_);
    float v = 0.0f;
    if (col < S_) {
        float t = tanhf(Z[i] * scale[col] + shift[col]);
        if (use_skip) t += H[i] * diag[col];
        v = t;
    }
    H[i] = v;
    Abf[i] = f2bf(v);
}

// ---------------------------------------------------------------------------
// out[b] = sum_n h[b,n]*fasm[n]*w_out[n] + b_out
// ---------------------------------------------------------------------------
__global__ __launch_bounds__(256) void head_kernel(
    const float* __restrict__ H, const float* __restrict__ fasm,
    const float* __restrict__ wout, const float* __restrict__ bout,
    float* __restrict__ out)
{
    __shared__ float red[256];
    const long row = blockIdx.x;
    float s = 0.0f;
    for (int c = threadIdx.x; c < S_; c += 256)
        s += H[row * SP_ + c] * fasm[c] * wout[c];
    red[threadIdx.x] = s;
    __syncthreads();
    for (int w = 128; w > 0; w >>= 1) {
        if (threadIdx.x < w) red[threadIdx.x] += red[threadIdx.x + w];
        __syncthreads();
    }
    if (threadIdx.x == 0) out[row] = red[0] + bout[0];
}

// ---------------------------------------------------------------------------
// Launch
// ---------------------------------------------------------------------------
extern "C" void kernel_launch(void* const* d_in, const int* in_sizes, int n_in,
                              void* d_out, int out_size, void* d_ws, size_t ws_size,
                              hipStream_t stream) {
    (void)in_sizes; (void)n_in; (void)out_size; (void)ws_size;

    const float* x     = (const float*)d_in[0];
    const float* lm0   = (const float*)d_in[1];
    const float* lm    = (const float*)d_in[2];
    const float* flm   = (const float*)d_in[3];
    const float* fasm  = (const float*)d_in[4];
    const float* W0    = (const float*)d_in[5];
    /* d_in[6] = b0 : cancels inside BN mean subtraction */
    const float* W     = (const float*)d_in[7];
    /* d_in[8] = b  : cancels inside BN mean subtraction */
    const float* gamma = (const float*)d_in[9];
    const float* beta  = (const float*)d_in[10];
    const float* wout  = (const float*)d_in[11];
    const float* bout  = (const float*)d_in[12];
    float* out = (float*)d_out;

    // ---- workspace carve-up (256B aligned) ----
    char* ws = (char*)d_ws;
    size_t off = 0;
    auto carve = [&](size_t bytes) -> char* {
        char* p = ws + off;
        off = (off + bytes + 255) & ~(size_t)255;
        return p;
    };
    unsigned short* Wt0 = (unsigned short*)carve((size_t)SP_ * KP0_ * 2);         // [SP, KP0]
    unsigned short* WtL = (unsigned short*)carve((size_t)L_ * SP_ * SP_ * 2);     // [L, SP, SP]
    unsigned short* A0  = (unsigned short*)carve((size_t)BN_ * KP0_ * 2);         // [B, KP0]
    unsigned short* Abf = (unsigned short*)carve((size_t)BN_ * SP_ * 2);          // [B, SP]
    float* Z      = (float*)carve((size_t)BN_ * SP_ * 4);                         // [B, SP]
    float* H      = (float*)carve((size_t)BN_ * SP_ * 4);                         // [B, SP]
    float* psum   = (float*)carve((size_t)(BN_ / 256) * SP_ * 4);
    float* psumsq = (float*)carve((size_t)(BN_ / 256) * SP_ * 4);
    float* scale  = (float*)carve((size_t)SP_ * 4);
    float* shift  = (float*)carve((size_t)SP_ * 4);

    // ---- weight prep: mask, transpose to [N,K], cast bf16 (L2-resident later)
    prep_weights_kernel<<<dim3(KP0_ / 32, SP_ / 32, 1), 256, 0, stream>>>(
        W0, lm0, Wt0, G_, S_, S_, KP0_, 0L, 0L);
    prep_weights_kernel<<<dim3(SP_ / 32, SP_ / 32, L_), 256, 0, stream>>>(
        W, lm, WtL, S_, S_, S_, SP_, (long)S_ * S_, (long)SP_ * SP_);

    // ---- depth-0 activation prep
    {
        long tot = (long)BN_ * KP0_;
        prep_a0_kernel<<<(unsigned)((tot + 255) / 256), 256, 0, stream>>>(x, A0, tot);
    }

    const dim3 gemmGrid(SP_ / 64, BN_ / 256);
    const long totBS = (long)BN_ * SP_;
    const unsigned applyBlocks = (unsigned)((totBS + 255) / 256);

    // ---- depth 0: masked linear -> BN -> tanh (no skip)
    gemm_bf16_kernel<<<gemmGrid, 256, 0, stream>>>(A0, Wt0, Z, psum, psumsq, KP0_, SP_);
    stats_finalize_kernel<<<SP_ / 256, 256, 0, stream>>>(
        psum, psumsq, gamma + 0 * S_, beta + 0 * S_, scale, shift, SP_);
    apply_kernel<<<applyBlocks, 256, 0, stream>>>(
        Z, H, Abf, scale, shift, flm /*unused*/, 0, totBS);

    // ---- depths 1..6: masked linear -> BN -> tanh + diagonal skip
    for (int l = 0; l < L_; ++l) {
        gemm_bf16_kernel<<<gemmGrid, 256, 0, stream>>>(
            Abf, WtL + (size_t)l * SP_ * SP_, Z, psum, psumsq, SP_, SP_);
        stats_finalize_kernel<<<SP_ / 256, 256, 0, stream>>>(
            psum, psumsq, gamma + (size_t)(l + 1) * S_, beta + (size_t)(l + 1) * S_,
            scale, shift, SP_);
        apply_kernel<<<applyBlocks, 256, 0, stream>>>(
            Z, H, Abf, scale, shift, flm + (size_t)l * S_, 1, totBS);
    }

    // ---- head
    head_kernel<<<BN_, 256, 0, stream>>>(H, fasm, wout, bout, out);
}
